// SGCRN_RD_22325240005090
// MI455X (gfx1250) — matrix operand order repaired and checked by reading, not compiled, not run on hardware
//
#include <hip/hip_runtime.h>
#include <hip/hip_bf16.h>

// ---------------------------------------------------------------------------
// SGCRN on MI455X (gfx1250). All heavy GEMMs run on V_WMMA_F32_16X16X32_BF16.
#define NN 512   // nodes
#define TT 12    // timesteps
#define BB 64    // batch
#define HH 64    // hidden
#define CC 65    // D_IN + H
#define RR 160   // padded gconv K-rows (2*CC=130 -> 160 = 5 * 32)
#define GG 128   // 2*H gate output
#define DE 10    // node-embedding dim
#define MN 20    // memory slots
#define HOR 12   // horizon
#define JJ (BB * CC)       // dense diffusion N-dim = 4160 (260 full 16-tiles)
#define NTILE (JJ / 16)    // 260 (divisible by 4 -> 65 wave-groups of 4 tiles)
// ---------------------------------------------------------------------------

typedef __attribute__((ext_vector_type(16))) __bf16 v16bf;
typedef __attribute__((ext_vector_type(8)))  __bf16 v8bf;
typedef __attribute__((ext_vector_type(8)))  float  v8f;

union AFrag { v16bf v; v8bf h[2]; };

static __device__ __forceinline__ float sigf(float x) {
  return 1.0f / (1.0f + __expf(-x));
}

// ---------------------------------------------------------------------------
// One-time: A = softmax(relu(E @ E^T), axis=1), stored bf16 (512x512).
// ---------------------------------------------------------------------------
__global__ void k_supports(const float* __restrict__ E, __bf16* __restrict__ A) {
  __shared__ float red[256];
  __shared__ float rowstat;
  int n = blockIdx.x, tid = threadIdx.x;
  float en[DE];
#pragma unroll
  for (int d = 0; d < DE; ++d) en[d] = E[n * DE + d];
  float v0 = 0.f, v1 = 0.f;
#pragma unroll
  for (int d = 0; d < DE; ++d) {
    v0 += en[d] * E[(tid) * DE + d];
    v1 += en[d] * E[(tid + 256) * DE + d];
  }
  v0 = fmaxf(v0, 0.f); v1 = fmaxf(v1, 0.f);
  red[tid] = fmaxf(v0, v1); __syncthreads();
  for (int s = 128; s > 0; s >>= 1) {
    if (tid < s) red[tid] = fmaxf(red[tid], red[tid + s]);
    __syncthreads();
  }
  if (tid == 0) rowstat = red[0];
  __syncthreads();
  float mx = rowstat;
  float e0 = __expf(v0 - mx), e1 = __expf(v1 - mx);
  red[tid] = e0 + e1; __syncthreads();
  for (int s = 128; s > 0; s >>= 1) {
    if (tid < s) red[tid] += red[tid + s];
    __syncthreads();
  }
  if (tid == 0) rowstat = red[0];
  __syncthreads();
  float inv = 1.0f / rowstat;
  A[(size_t)n * NN + tid]       = (__bf16)(e0 * inv);
  A[(size_t)n * NN + tid + 256] = (__bf16)(e1 * inv);
}

// ---------------------------------------------------------------------------
// Fold node embeddings into per-node weights: out[n][r][o] bf16, r in [0,160),
// r = k*65 + c for r < 130, zero-padded above. W layout (DE, 2, CC, O).
// ---------------------------------------------------------------------------
__global__ void k_mkw(const float* __restrict__ E, const float* __restrict__ W,
                      __bf16* __restrict__ out, int O, int total) {
  int id = blockIdx.x * blockDim.x + threadIdx.x;
  if (id >= total) return;
  int o = id % O;
  int r = (id / O) % RR;
  int n = id / (O * RR);
  float acc = 0.f;
  if (r < 2 * CC) {
    int k = r / CC, c = r % CC;
    const float* wp = W + ((size_t)k * CC + c) * O + o;
#pragma unroll
    for (int d = 0; d < DE; ++d)
      acc += E[n * DE + d] * wp[(size_t)d * 2 * CC * O];
  }
  out[id] = (__bf16)acc;
}

// Fold biases: out[n][o] = sum_d E[n,d] * b[d,o]
__global__ void k_mkb(const float* __restrict__ E, const float* __restrict__ bsrc,
                      float* __restrict__ out, int O) {
  int id = blockIdx.x * blockDim.x + threadIdx.x;
  if (id >= NN * O) return;
  int o = id % O, n = id / O;
  float acc = 0.f;
#pragma unroll
  for (int d = 0; d < DE; ++d) acc += E[n * DE + d] * bsrc[d * O + o];
  out[id] = acc;
}

__global__ void k_fill0(float* __restrict__ p, int total) {
  int i = blockIdx.x * blockDim.x + threadIdx.x;
  if (i < total) p[i] = 0.f;
}

// src1 accumulator init: acc[b][o][n] = conv2_b[o]
__global__ void k_init_acc(const float* __restrict__ c2b, float* __restrict__ acc) {
  int i = blockIdx.x * blockDim.x + threadIdx.x;
  if (i >= BB * HOR * NN) return;
  int o = (i / NN) % HOR;
  acc[i] = c2b[o];
}

// ---------------------------------------------------------------------------
// Build concat + dense buffers for one timestep.
//   xsg[n][b][0]=x_t, [1..64]=h, [130..159]=0 ; xcg row0=x_t, pads=0.
//   xd [n][b*65 + c] = same xs values (dense GEMM operand for diffusion).
// xsrc indexed as [b][t][n] (matches both `source` (D_IN=1) and residual).
// ---------------------------------------------------------------------------
__global__ void k_build_xs(const float* __restrict__ xsrc, int t,
                           const float* __restrict__ h,
                           __bf16* __restrict__ xsg, __bf16* __restrict__ xcg,
                           __bf16* __restrict__ xd) {
  int n = blockIdx.x;
  for (int idx = threadIdx.x; idx < BB * RR; idx += blockDim.x) {
    int b = idx / RR, col = idx % RR;
    size_t off = ((size_t)n * BB + b) * RR + col;
    if (col == 0) {
      float x = xsrc[((size_t)b * TT + t) * NN + n];
      xsg[off] = (__bf16)x;
      xcg[off] = (__bf16)x;
      xd[(size_t)n * JJ + b * CC] = (__bf16)x;
    } else if (col < CC) {
      __bf16 hv = (__bf16)h[((size_t)n * BB + b) * HH + (col - 1)];
      xsg[off] = hv;
      xd[(size_t)n * JJ + b * CC + col] = hv;
    } else if (col >= 2 * CC) {
      xsg[off] = (__bf16)0.f;
      xcg[off] = (__bf16)0.f;
    }
  }
}

// ---------------------------------------------------------------------------
// Diffusion GEMM (WMMA bf16): Y[n][b][65+c] = sum_m A[n][m] * Xd[m][b*65+c].
// M=512 (nodes), N=4160 dense columns, K=512.
// Block = 8 waves, owns one 16-node A panel staged in LDS (16x512 bf16, 16KB);
// each wave computes 4 adjacent 16x16 N-tiles so one LDS A-fragment feeds
// 4 WMMAs per k-step.  260 tiles = 65 wave-groups of exactly 4 -> a wave is
// either fully active or fully idle; idle waves exit via a SCALAR branch
// (readfirstlane) so all executed WMMAs see EXEC == all-ones, as the ISA
// requires.  Hot loop uses pointer stepping: loads + WMMA + one 64-bit add.
// ---------------------------------------------------------------------------
__global__ void k_diffuse(const __bf16* __restrict__ A,
                          const __bf16* __restrict__ Xd,
                          __bf16* __restrict__ Y) {
  __shared__ __bf16 lA[16 * NN];  // 16 rows x 512 cols
  int tid = threadIdx.x;
  int tm = blockIdx.x & 31;       // node tile
  int ng = blockIdx.x >> 5;       // N-group of 32 tiles

  // Cooperative A-panel load: 8192 bf16, 32 per thread (4 x b128).
  {
    int start = tid * 32;
    int row = start >> 9, col = start & 511;
    const __bf16* gsrc = A + (size_t)(tm * 16 + row) * NN + col;
    __bf16* ldst = lA + row * NN + col;
#pragma unroll
    for (int q = 0; q < 4; ++q)
      *(v8bf*)(ldst + q * 8) = *(const v8bf*)(gsrc + q * 8);
  }
  __syncthreads();

  int lane = tid & 31, w = tid >> 5;
  // Wave-uniform tile base, forced into an SGPR -> scalar early-out.
  int base_tn = __builtin_amdgcn_readfirstlane(ng * 32 + w * 4);
  if (base_tn >= NTILE) return;   // scalar branch; no EXEC masking downstream

  int r = lane & 15, half = lane >> 4;
  const __bf16* lptr = lA + r * NN + half * 8;              // LDS A-frag ptr
  const __bf16* brow = Xd + (size_t)(half * 8) * JJ + base_tn * 16 + r;  // B col ptr
  v8f acc0 = {}, acc1 = {}, acc2 = {}, acc3 = {};

  for (int k0 = 0; k0 < NN; k0 += 32) {
    AFrag a;
    a.h[0] = *(const v8bf*)(lptr);        // ds_load_b128
    a.h[1] = *(const v8bf*)(lptr + 16);
    __builtin_prefetch(brow + 32 * JJ, 0, 1);  // global_prefetch_b8
    AFrag b0, b1, b2, b3;
#pragma unroll
    for (int s = 0; s < 16; ++s) {
      const __bf16* bp = brow + (size_t)((s & 7) + (s >> 3) * 16) * JJ;
      b0.v[s] = bp[0]; b1.v[s] = bp[16]; b2.v[s] = bp[32]; b3.v[s] = bp[48];
    }
    acc0 = __builtin_amdgcn_wmma_f32_16x16x32_bf16(
        false, a.v, false, b0.v, (short)0, acc0, false, false);
    acc1 = __builtin_amdgcn_wmma_f32_16x16x32_bf16(
        false, a.v, false, b1.v, (short)0, acc1, false, false);
    acc2 = __builtin_amdgcn_wmma_f32_16x16x32_bf16(
        false, a.v, false, b2.v, (short)0, acc2, false, false);
    acc3 = __builtin_amdgcn_wmma_f32_16x16x32_bf16(
        false, a.v, false, b3.v, (short)0, acc3, false, false);
    lptr += 32;
    brow += (size_t)32 * JJ;
  }

  v8f accs[4] = {acc0, acc1, acc2, acc3};
#pragma unroll
  for (int i = 0; i < 4; ++i) {
    int j = (base_tn + i) * 16 + (lane & 15);
    int bb = j / CC, cc = j - bb * CC;
#pragma unroll
    for (int e = 0; e < 8; ++e) {
      int node = tm * 16 + e + half * 8;
      Y[((size_t)node * BB + bb) * RR + CC + cc] = (__bf16)accs[i][e];
    }
  }
}

// ---------------------------------------------------------------------------
// Per-node gate GEMM (WMMA bf16): Out(64x128) = Act(64x160) @ Wg[n](160x128),
// sigmoid; cols [0,64) -> z: write z*h into xcg rows 1..64 AND dense xd;
// cols [64,128) -> r (stash in rbuf).  One block (8 waves) per node.
// ---------------------------------------------------------------------------
__global__ void k_gate(const __bf16* __restrict__ xsg, const __bf16* __restrict__ Wg,
                       const float* __restrict__ bg, const float* __restrict__ h,
                       __bf16* __restrict__ xcg, __bf16* __restrict__ xd,
                       float* __restrict__ rbuf) {
  int node = blockIdx.x;
  int lane = threadIdx.x & 31, w = threadIdx.x >> 5;
  const __bf16* act = xsg + (size_t)node * BB * RR;
  const __bf16* wgt = Wg + (size_t)node * RR * GG;
  int r = lane & 15, half = lane >> 4;
  for (int i = 0; i < 4; ++i) {
    int tidx = w + 8 * i;          // 0..31
    int tm = tidx >> 3, tn = tidx & 7;
    const __bf16* aptr = act + (size_t)(tm * 16 + r) * RR + half * 8;
    const __bf16* bptr = wgt + (size_t)(half * 8) * GG + tn * 16 + r;
    v8f acc = {};
#pragma unroll
    for (int k0 = 0; k0 < RR; k0 += 32) {
      AFrag a, b;
      a.h[0] = *(const v8bf*)(aptr);
      a.h[1] = *(const v8bf*)(aptr + 16);
#pragma unroll
      for (int s = 0; s < 16; ++s) b.v[s] = bptr[(size_t)((s & 7) + (s >> 3) * 16) * GG];
      acc = __builtin_amdgcn_wmma_f32_16x16x32_bf16(
          false, a.v, false, b.v, (short)0, acc, false, false);
      aptr += 32;
      bptr += 32 * GG;
    }
#pragma unroll
    for (int e = 0; e < 8; ++e) {
      int b_ = tm * 16 + e + half * 8;
      int o  = tn * 16 + (lane & 15);
      float s = sigf(acc[e] + bg[(size_t)node * GG + o]);
      if (o < HH) {
        float hv = h[((size_t)node * BB + b_) * HH + o];
        __bf16 zh = (__bf16)(s * hv);
        xcg[((size_t)node * BB + b_) * RR + 1 + o] = zh;
        xd[(size_t)node * JJ + b_ * CC + 1 + o] = zh;
      } else {
        rbuf[((size_t)node * BB + b_) * HH + (o - HH)] = s;
      }
    }
  }
}

// ---------------------------------------------------------------------------
// Per-node update GEMM (WMMA bf16): hc = tanh(Act(64x160) @ Wu[n](160x64)+bu),
// then GRU blend h = r*h + (1-r)*hc in place.
// ---------------------------------------------------------------------------
__global__ void k_upd(const __bf16* __restrict__ xcg, const __bf16* __restrict__ Wu,
                      const float* __restrict__ bu, const float* __restrict__ rbuf,
                      float* __restrict__ h) {
  int node = blockIdx.x;
  int lane = threadIdx.x & 31, w = threadIdx.x >> 5;
  const __bf16* act = xcg + (size_t)node * BB * RR;
  const __bf16* wgt = Wu + (size_t)node * RR * HH;
  int r = lane & 15, half = lane >> 4;
  for (int i = 0; i < 2; ++i) {
    int tidx = w + 8 * i;          // 0..15
    int tm = tidx >> 2, tn = tidx & 3;
    const __bf16* aptr = act + (size_t)(tm * 16 + r) * RR + half * 8;
    const __bf16* bptr = wgt + (size_t)(half * 8) * HH + tn * 16 + r;
    v8f acc = {};
#pragma unroll
    for (int k0 = 0; k0 < RR; k0 += 32) {
      AFrag a, b;
      a.h[0] = *(const v8bf*)(aptr);
      a.h[1] = *(const v8bf*)(aptr + 16);
#pragma unroll
      for (int s = 0; s < 16; ++s) b.v[s] = bptr[(size_t)((s & 7) + (s >> 3) * 16) * HH];
      acc = __builtin_amdgcn_wmma_f32_16x16x32_bf16(
          false, a.v, false, b.v, (short)0, acc, false, false);
      aptr += 32;
      bptr += 32 * HH;
    }
#pragma unroll
    for (int e = 0; e < 8; ++e) {
      int b_ = tm * 16 + e + half * 8;
      int j  = tn * 16 + (lane & 15);
      float hc = tanhf(acc[e] + bu[(size_t)node * HH + j]);
      size_t hidx = ((size_t)node * BB + b_) * HH + j;
      float rv = rbuf[hidx];
      h[hidx] = rv * h[hidx] + (1.0f - rv) * hc;
    }
  }
}

// Incremental conv2 einsum: acc[b][o][n] += sum_j h[n][b][j]*conv2_W[o][j][t]
__global__ void k_accum(const float* __restrict__ h, const float* __restrict__ c2w,
                        float* __restrict__ acc, int t) {
  int n = blockIdx.x, b = threadIdx.x;  // 64 threads
  const float* hp = h + ((size_t)n * BB + b) * HH;
  float th[HH];
#pragma unroll
  for (int j = 0; j < HH; ++j) th[j] = hp[j];
  for (int o = 0; o < HOR; ++o) {
    float s = 0.f;
#pragma unroll
    for (int j = 0; j < HH; ++j) s += th[j] * c2w[((size_t)o * HH + j) * HOR + t];
    acc[((size_t)b * HOR + o) * NN + n] += s;
  }
}

// residual[b][t][n] = source[b][t][n] - src1[b][t][n]  (identical flat layout)
__global__ void k_residual(const float* __restrict__ src, const float* __restrict__ acc,
                           float* __restrict__ res) {
  int i = blockIdx.x * blockDim.x + threadIdx.x;
  if (i < BB * TT * NN) res[i] = src[i] - acc[i];
}

// Memory attention + proj + conv1 -> writes d_out (= trend_out).
__global__ void k_trend(const float* __restrict__ h, const float* __restrict__ mem,
                        const float* __restrict__ Wq, const float* __restrict__ mw,
                        const float* __restrict__ projW, const float* __restrict__ projb,
                        const float* __restrict__ c1w, const float* __restrict__ c1b,
                        float* __restrict__ out) {
  int n = blockIdx.x, b = threadIdx.x;
  const float* hp = h + ((size_t)n * BB + b) * HH;
  float th[HH], q[HH];
#pragma unroll
  for (int j = 0; j < HH; ++j) th[j] = hp[j];
  for (int j = 0; j < HH; ++j) {
    float s = 0.f;
    for (int k = 0; k < HH; ++k) s += th[k] * Wq[j * HH + k];
    q[j] = s;
  }
  float logit[MN], mx = -3.0e38f;
  for (int m = 0; m < MN; ++m) {
    float s = 0.f;
    for (int j = 0; j < HH; ++j) s += q[j] * mem[m * HH + j];
    logit[m] = s; mx = fmaxf(mx, s);
  }
  float sum = 0.f;
  for (int m = 0; m < MN; ++m) { logit[m] = __expf(logit[m] - mx); sum += logit[m]; }
  float inv = 1.0f / sum;
  float wv = mw[0];
  float he[HH];
  for (int j = 0; j < HH; ++j) {
    float ha = 0.f;
    for (int m = 0; m < MN; ++m) ha += logit[m] * mem[m * HH + j];
    he[j] = wv * (ha * inv) + (1.0f - wv) * th[j];
  }
  float pe[HH];
  for (int j = 0; j < HH; ++j) {
    float s = projb[j];
    for (int k = 0; k < HH; ++k) s += he[k] * projW[j * HH + k];
    pe[j] = s;
  }
  for (int o = 0; o < HOR; ++o) {
    float s = c1b[o];
    for (int j = 0; j < HH; ++j) s += pe[j] * c1w[o * HH + j];
    out[((size_t)b * HOR + o) * NN + n] = s;
  }
}

// residual_out added into d_out: out[b][o][n] += h2[n][b][:] . conv3_W[o][:] + b
__global__ void k_final(const float* __restrict__ h, const float* __restrict__ c3w,
                        const float* __restrict__ c3b, float* __restrict__ out) {
  int n = blockIdx.x, b = threadIdx.x;
  const float* hp = h + ((size_t)n * BB + b) * HH;
  float th[HH];
#pragma unroll
  for (int j = 0; j < HH; ++j) th[j] = hp[j];
  for (int o = 0; o < HOR; ++o) {
    float s = c3b[o];
#pragma unroll
    for (int j = 0; j < HH; ++j) s += th[j] * c3w[o * HH + j];
    out[((size_t)b * HOR + o) * NN + n] += s;
  }
}

// ---------------------------------------------------------------------------
extern "C" void kernel_launch(void* const* d_in, const int* in_sizes, int n_in,
                              void* d_out, int out_size, void* d_ws, size_t ws_size,
                              hipStream_t stream) {
  (void)in_sizes; (void)n_in; (void)out_size; (void)ws_size;
  const float* src    = (const float*)d_in[0];
  const float* E      = (const float*)d_in[1];
  const float* gW[2]  = {(const float*)d_in[2], (const float*)d_in[6]};
  const float* gB[2]  = {(const float*)d_in[3], (const float*)d_in[7]};
  const float* uW[2]  = {(const float*)d_in[4], (const float*)d_in[8]};
  const float* uB[2]  = {(const float*)d_in[5], (const float*)d_in[9]};
  const float* memory = (const float*)d_in[10];
  const float* Wq     = (const float*)d_in[11];
  const float* mw     = (const float*)d_in[12];
  const float* projW  = (const float*)d_in[13];
  const float* projb  = (const float*)d_in[14];
  const float* c1w    = (const float*)d_in[15];
  const float* c1b    = (const float*)d_in[16];
  const float* c2w    = (const float*)d_in[17];
  const float* c2b    = (const float*)d_in[18];
  const float* c3w    = (const float*)d_in[19];
  const float* c3b    = (const float*)d_in[20];
  float* out = (float*)d_out;

  // Workspace carve-up (256B aligned).
  char* ws = (char*)d_ws;
  size_t off = 0;
  auto alloc = [&](size_t bytes) {
    char* p = ws + off;
    off += (bytes + 255) & ~(size_t)255;
    return p;
  };
  __bf16* A_bf   = (__bf16*)alloc((size_t)NN * NN * 2);
  __bf16* Wg_bf[2], *Wu_bf[2];
  float *bg_f[2], *bu_f[2];
  for (int e = 0; e < 2; ++e) {
    Wg_bf[e] = (__bf16*)alloc((size_t)NN * RR * GG * 2);
    Wu_bf[e] = (__bf16*)alloc((size_t)NN * RR * HH * 2);
    bg_f[e]  = (float*)alloc((size_t)NN * GG * 4);
    bu_f[e]  = (float*)alloc((size_t)NN * HH * 4);
  }
  float*  hbuf = (float*)alloc((size_t)NN * BB * HH * 4);
  float*  rbuf = (float*)alloc((size_t)NN * BB * HH * 4);
  __bf16* xsg  = (__bf16*)alloc((size_t)NN * BB * RR * 2);
  __bf16* xcg  = (__bf16*)alloc((size_t)NN * BB * RR * 2);
  __bf16* xd   = (__bf16*)alloc((size_t)NN * JJ * 2);   // dense diffusion operand
  float*  acc  = (float*)alloc((size_t)BB * HOR * NN * 4);
  float*  res  = (float*)alloc((size_t)BB * TT * NN * 4);

  // ---- One-time folds ----
  k_supports<<<NN, 256, 0, stream>>>(E, A_bf);
  for (int e = 0; e < 2; ++e) {
    int tg = NN * RR * GG;
    k_mkw<<<(tg + 255) / 256, 256, 0, stream>>>(E, gW[e], Wg_bf[e], GG, tg);
    int tu = NN * RR * HH;
    k_mkw<<<(tu + 255) / 256, 256, 0, stream>>>(E, uW[e], Wu_bf[e], HH, tu);
    k_mkb<<<(NN * GG + 255) / 256, 256, 0, stream>>>(E, gB[e], bg_f[e], GG);
    k_mkb<<<(NN * HH + 255) / 256, 256, 0, stream>>>(E, uB[e], bu_f[e], HH);
  }
  k_init_acc<<<(BB * HOR * NN + 255) / 256, 256, 0, stream>>>(c2b, acc);

  const int diffBlocks = 32 * ((NTILE + 31) / 32);  // 32 M-tiles x 9 N-groups

  for (int enc = 0; enc < 2; ++enc) {
    const float* xin = (enc == 0) ? src : res;
    k_fill0<<<(NN * BB * HH + 255) / 256, 256, 0, stream>>>(hbuf, NN * BB * HH);
    for (int t = 0; t < TT; ++t) {
      k_build_xs<<<NN, 256, 0, stream>>>(xin, t, hbuf, xsg, xcg, xd);
      k_diffuse<<<diffBlocks, 256, 0, stream>>>(A_bf, xd, xsg);
      k_gate<<<NN, 256, 0, stream>>>(xsg, Wg_bf[enc], bg_f[enc], hbuf, xcg, xd, rbuf);
      k_diffuse<<<diffBlocks, 256, 0, stream>>>(A_bf, xd, xcg);
      k_upd<<<NN, 256, 0, stream>>>(xcg, Wu_bf[enc], bu_f[enc], rbuf, hbuf);
      if (enc == 0) k_accum<<<NN, BB, 0, stream>>>(hbuf, c2w, acc, t);
    }
    if (enc == 0) {
      k_trend<<<NN, BB, 0, stream>>>(hbuf, memory, Wq, mw, projW, projb, c1w, c1b, out);
      k_residual<<<(BB * TT * NN + 255) / 256, 256, 0, stream>>>(src, acc, res);
    } else {
      k_final<<<NN, BB, 0, stream>>>(hbuf, c3w, c3b, out);
    }
  }
}
